// MultiHeadAttention_27393301414082
// MI455X (gfx1250) — matrix operand order repaired and checked by reading, not compiled, not run on hardware
//
#include <hip/hip_runtime.h>

typedef __attribute__((ext_vector_type(16))) _Float16 v16h;
typedef __attribute__((ext_vector_type(8)))  _Float16 v8h;
typedef __attribute__((ext_vector_type(8)))  float    v8f;
typedef __attribute__((vector_size(16)))     int      b128_t;

#define B_    8
#define L_    2048
#define E_    512
#define H_    8
#define D_    64
#define NROWS (B_*H_*L_)                 // 131072 rows of 64, same for q/k/v views
#define SCALE 0.04419417382415922f       // 1/sqrt(512)

#define AS1 __attribute__((address_space(1)))
#define AS3 __attribute__((address_space(3)))

#if __has_builtin(__builtin_amdgcn_global_load_async_to_lds_b128)
#define HAVE_ASYNC 1
#else
#define HAVE_ASYNC 0
#endif

static __device__ __forceinline__ v8f wmma16(v16h a, v16h b, v8f c) {
  // v_wmma_f32_16x16x32_f16: D = A(16x32 f16) * B(32x16 f16) + C(16x16 f32)
  return __builtin_amdgcn_wmma_f32_16x16x32_f16(false, a, false, b, (short)0, c, false, false);
}

static __device__ __forceinline__ void wait_async0() {
#if __has_builtin(__builtin_amdgcn_s_wait_asynccnt)
  __builtin_amdgcn_s_wait_asynccnt(0);
#else
  asm volatile("s_wait_asynccnt 0x0" ::: "memory");
#endif
}

#if HAVE_ASYNC
static __device__ __forceinline__ void async_copy16(const _Float16* g, _Float16* l) {
  // GLOBAL_LOAD_ASYNC_TO_LDS_B128: per-lane 16B global->LDS, tracked by ASYNCcnt
  __builtin_amdgcn_global_load_async_to_lds_b128(
      (AS1 b128_t*)(void*)g, (AS3 b128_t*)(void*)l, 0, 0);
}
#endif

// ---------------------------------------------------------------------------
// Kernel 1: q/k/v projections.  Y(131072x64, f16) = X(131072x64, f32) @ W^T.
// blockIdx.y selects which of the three projections; 8 waves x 16 rows each.
// ---------------------------------------------------------------------------
__global__ void __launch_bounds__(256) proj_qkv(
    const float* __restrict__ q, const float* __restrict__ k, const float* __restrict__ v,
    const float* __restrict__ Wq, const float* __restrict__ Wk, const float* __restrict__ Wv,
    _Float16* __restrict__ qp, _Float16* __restrict__ kp, _Float16* __restrict__ vp)
{
  const float* X; const float* W; _Float16* Y;
  if (blockIdx.y == 0)      { X = q; W = Wq; Y = qp; }
  else if (blockIdx.y == 1) { X = k; W = Wk; Y = kp; }
  else                      { X = v; W = Wv; Y = vp; }

  __shared__ _Float16 sW[64 * 64];        // W row-major, f16
  int t = threadIdx.x;
  for (int i = t; i < 1024; i += 256) {
    float4 w4 = ((const float4*)W)[i];
    sW[4*i+0] = (_Float16)w4.x; sW[4*i+1] = (_Float16)w4.y;
    sW[4*i+2] = (_Float16)w4.z; sW[4*i+3] = (_Float16)w4.w;
  }
  __syncthreads();

  int wave = t >> 5, lane = t & 31;
  int lr = lane & 15, ch = lane >> 4, ks = ch * 8;
  int rowBase = blockIdx.x * 128 + wave * 16;

  // A fragments: row = rowBase + lr, K = 0..63 (two 16x32 fragments)
  const float* xr = X + (size_t)(rowBase + lr) * 64;
  v16h a[2];
  #pragma unroll
  for (int kb = 0; kb < 2; ++kb) {
    #pragma unroll
    for (int j = 0; j < 8; ++j) a[kb][j]     = (_Float16)xr[32*kb + ks + j];
    #pragma unroll
    for (int j = 0; j < 8; ++j) a[kb][8 + j] = (_Float16)xr[32*kb + 16 + ks + j];
  }

  #pragma unroll
  for (int nt = 0; nt < 4; ++nt) {
    int n = nt * 16 + lr;                 // output column; B[k][n] = W[n][k]
    v8f c = {};
    #pragma unroll
    for (int kb = 0; kb < 2; ++kb) {
      v16h bf;
      const _Float16* wr = sW + n * 64 + kb * 32 + ch * 8;
      #pragma unroll
      for (int j = 0; j < 16; ++j) bf[j] = wr[j + (j & 8)];
      c = wmma16(a[kb], bf, c);
    }
    #pragma unroll
    for (int i = 0; i < 8; ++i)
      Y[(size_t)(rowBase + i + 8*ch) * 64 + n] = (_Float16)c[i];
  }
}

// ---------------------------------------------------------------------------
// Kernel 2: flash attention.  grid = (L/128, B*H).  Each wave owns 16 query
// rows.  K tiles are double-buffered and staged with async global->LDS copies
// (ASYNCcnt); V tiles are staged transposed through registers.
// ---------------------------------------------------------------------------
#define TK   64
#define KPAD 72   // 72 halves = 144 B row pitch -> conflict-free ds_load_b128
#define NT_  (L_ / TK)

__global__ void __launch_bounds__(256) flash_attn(
    const _Float16* __restrict__ qp, const _Float16* __restrict__ kp,
    const _Float16* __restrict__ vp, _Float16* __restrict__ ao)
{
  __shared__ _Float16 sK[2][64 * KPAD];       // K tile x2 (double buffer), row-major
  __shared__ _Float16 sV[64 * KPAD];          // V tile, transposed (dim x seq)
  __shared__ _Float16 sP[8 * 16 * KPAD];      // per-wave P strip (C -> A relayout)

  int t = threadIdx.x;
  int wave = t >> 5, lane = t & 31, lr = lane & 15, ch = lane >> 4, ks = ch * 8;
  int bh = blockIdx.y;                         // b*8 + h
  int b  = bh >> 3, h = bh & 7;
  int qTile = blockIdx.x * 128 + wave * 16;

  // Q fragments (16 rows x 64 K), softmax scale folded in
  const _Float16* qrow = qp + ((size_t)bh * L_ + qTile + lr) * 64;
  v16h aq[2];
  #pragma unroll
  for (int kb = 0; kb < 2; ++kb) {
    #pragma unroll
    for (int j = 0; j < 8; ++j) aq[kb][j]     = qrow[32*kb + ks + j]      * (_Float16)SCALE;
    #pragma unroll
    for (int j = 0; j < 8; ++j) aq[kb][8 + j] = qrow[32*kb + 16 + ks + j] * (_Float16)SCALE;
  }

  // issue K-tile staging (async path when available): 512 chunks of 16B
  auto issueK = [&](int kt, int buf) {
    const _Float16* kbase = kp + ((size_t)bh * L_ + kt * 64) * 64;
    #pragma unroll
    for (int c0 = 0; c0 < 2; ++c0) {
      int c = t + c0 * 256;
      int row = c >> 3, off = (c & 7) * 8;
#if HAVE_ASYNC
      async_copy16(kbase + row * 64 + off, &sK[buf][row * KPAD + off]);
#else
      v8h kv = *(const v8h*)(kbase + row * 64 + off);
      *(v8h*)(&sK[buf][row * KPAD + off]) = kv;
#endif
    }
  };

  float m[8], lsum[8];
  v8f o[4];
  v8f zf = {};
  #pragma unroll
  for (int i = 0; i < 8; ++i) { m[i] = -1e30f; lsum[i] = 0.f; }
  #pragma unroll
  for (int ct = 0; ct < 4; ++ct) o[ct] = zf;

  _Float16* sPw = sP + wave * 16 * KPAD;

  issueK(0, 0);                                // prologue: tile 0 in flight

  for (int kt = 0; kt < NT_; ++kt) {
    int cur = kt & 1;
    wait_async0();                             // my tile-kt chunks landed in LDS
    __syncthreads();                           // everyone's landed; prior-iter LDS reads done

    if (kt + 1 < NT_) issueK(kt + 1, cur ^ 1); // overlap next K tile with compute

    // stage V tile transposed: vp row = (b*L + lv)*8 + h
    #pragma unroll
    for (int c0 = 0; c0 < 2; ++c0) {
      int c = t + c0 * 256;
      int lv = c >> 3, coff = (c & 7) * 8;
      const _Float16* vrow = vp + ((size_t)(b * L_ + kt * 64 + lv) * 8 + h) * 64;
      v8h vv = *(const v8h*)(vrow + coff);
      #pragma unroll
      for (int j = 0; j < 8; ++j) sV[(coff + j) * KPAD + lv] = vv[j];
    }
    if (kt + 1 < NT_) {                        // global_prefetch_b8 of next V tile
      const _Float16* vnext =
          vp + ((size_t)(b * L_ + (kt + 1) * 64 + (t >> 2)) * 8 + h) * 64 + (t & 3) * 16;
      __builtin_prefetch(vnext, 0, 1);
    }
    __syncthreads();

    // S = (Q*scale) @ K^T : 4 column subtiles, 2 K-steps each
    v8f s[4];
    #pragma unroll
    for (int st = 0; st < 4; ++st) {
      int n = st * 16 + lr;                   // S column = K-tile row
      v8f c = zf;
      #pragma unroll
      for (int kb = 0; kb < 2; ++kb) {
        v16h bf;
        const _Float16* kr = &sK[cur][n * KPAD + kb * 32 + ch * 8];
        #pragma unroll
        for (int j = 0; j < 16; ++j) bf[j] = kr[j + (j & 8)];
        c = wmma16(aq[kb], bf, c);
      }
      s[st] = c;
    }

    // online softmax: per-lane slot i holds row (i + 8*ch)
    float mnew[8], alpha[8], rs[8];
    #pragma unroll
    for (int i = 0; i < 8; ++i) {
      float mx = fmaxf(fmaxf(s[0][i], s[1][i]), fmaxf(s[2][i], s[3][i]));
      mx = fmaxf(mx, __shfl_xor(mx, 1));
      mx = fmaxf(mx, __shfl_xor(mx, 2));
      mx = fmaxf(mx, __shfl_xor(mx, 4));
      mx = fmaxf(mx, __shfl_xor(mx, 8));
      mnew[i] = fmaxf(m[i], mx);
      alpha[i] = __expf(m[i] - mnew[i]);
      m[i] = mnew[i];
      rs[i] = 0.f;
    }
    #pragma unroll
    for (int st = 0; st < 4; ++st)
      #pragma unroll
      for (int i = 0; i < 8; ++i) {
        float p = __expf(s[st][i] - mnew[i]);
        s[st][i] = p;
        rs[i] += p;
      }
    #pragma unroll
    for (int i = 0; i < 8; ++i) {
      float r = rs[i];
      r += __shfl_xor(r, 1); r += __shfl_xor(r, 2);
      r += __shfl_xor(r, 4); r += __shfl_xor(r, 8);
      lsum[i] = lsum[i] * alpha[i] + r;
    }
    #pragma unroll
    for (int ct = 0; ct < 4; ++ct)
      #pragma unroll
      for (int i = 0; i < 8; ++i) o[ct][i] *= alpha[i];

    // P: C-layout -> row-major f16 in per-wave LDS strip (same-wave, in-order)
    #pragma unroll
    for (int st = 0; st < 4; ++st)
      #pragma unroll
      for (int i = 0; i < 8; ++i)
        sPw[(i + 8*ch) * KPAD + st * 16 + lr] = (_Float16)s[st][i];

    // reload P as A fragments, O += P @ V
    v16h ap[2];
    #pragma unroll
    for (int kb = 0; kb < 2; ++kb) {
      const _Float16* pr = sPw + lr * KPAD + kb * 32;
      #pragma unroll
      for (int j = 0; j < 8; ++j) ap[kb][j]     = pr[ks + j];
      #pragma unroll
      for (int j = 0; j < 8; ++j) ap[kb][8 + j] = pr[16 + ks + j];
    }
    #pragma unroll
    for (int ct = 0; ct < 4; ++ct) {
      int n = ct * 16 + lr;                   // output head-dim column
      #pragma unroll
      for (int kb = 0; kb < 2; ++kb) {
        v16h bf;
        const _Float16* vr = sV + n * KPAD + kb * 32 + ch * 8;
        #pragma unroll
        for (int j = 0; j < 16; ++j) bf[j] = vr[j + (j & 8)];
        o[ct] = wmma16(ap[kb], bf, o[ct]);
      }
    }
  }

  // normalize and store to (B, L, H*D) f16
  #pragma unroll
  for (int i = 0; i < 8; ++i) lsum[i] = 1.f / lsum[i];
  #pragma unroll
  for (int ct = 0; ct < 4; ++ct) {
    int col = h * 64 + ct * 16 + lr;
    #pragma unroll
    for (int i = 0; i < 8; ++i) {
      int qr = qTile + i + 8 * ch;
      ao[((size_t)(b * L_) + qr) * E_ + col] = (_Float16)(o[ct][i] * lsum[i]);
    }
  }
}

// ---------------------------------------------------------------------------
// Kernel 3: output projection.  out(16384x512 f32) = AO(f16) @ Wo^T + bo.
// grid = (16384/16, 512/128); 8 waves each own a 16-column tile.
// ---------------------------------------------------------------------------
#define APAD 520

__global__ void __launch_bounds__(256) out_proj(
    const _Float16* __restrict__ ao, const float* __restrict__ Wo,
    const float* __restrict__ bo, float* __restrict__ out)
{
  __shared__ _Float16 sA[16 * APAD];
  int t = threadIdx.x;
  int wave = t >> 5, lane = t & 31, lr = lane & 15, ch = lane >> 4, ks = ch * 8;
  int rowBase = blockIdx.x * 16;

  const _Float16* abase = ao + (size_t)rowBase * E_;
  #pragma unroll
  for (int c0 = 0; c0 < 4; ++c0) {
    int c = t + c0 * 256;                     // 1024 chunks of 8 halves
    int row = c >> 6, off = (c & 63) * 8;
    v8h av = *(const v8h*)(abase + row * E_ + off);
    *(v8h*)(sA + row * APAD + off) = av;
  }
  __syncthreads();

  int n = blockIdx.y * 128 + wave * 16 + lr;  // output column; B[k][n] = Wo[n][k]
  const float* wr = Wo + (size_t)n * E_;
  v8f acc = {};
  #pragma unroll 4
  for (int kb = 0; kb < 16; ++kb) {
    v16h a, bf;
    const _Float16* ar = sA + lr * APAD + kb * 32;
    #pragma unroll
    for (int j = 0; j < 8; ++j) a[j]     = ar[ks + j];
    #pragma unroll
    for (int j = 0; j < 8; ++j) a[8 + j] = ar[16 + ks + j];
    const float* wk = wr + kb * 32 + ch * 8;
    #pragma unroll
    for (int j = 0; j < 16; ++j) bf[j] = (_Float16)wk[j + (j & 8)];
    acc = wmma16(a, bf, acc);
  }
  float bias = bo[n];
  #pragma unroll
  for (int i = 0; i < 8; ++i)
    out[(size_t)(rowBase + i + 8*ch) * E_ + n] = acc[i] + bias;
}

// ---------------------------------------------------------------------------
extern "C" void kernel_launch(void* const* d_in, const int* in_sizes, int n_in,
                              void* d_out, int out_size, void* d_ws, size_t ws_size,
                              hipStream_t stream) {
  const float* q  = (const float*)d_in[0];
  const float* k  = (const float*)d_in[1];
  const float* v  = (const float*)d_in[2];
  const float* Wq = (const float*)d_in[3];
  const float* Wk = (const float*)d_in[4];
  const float* Wv = (const float*)d_in[5];
  const float* Wo = (const float*)d_in[6];
  const float* bo = (const float*)d_in[7];
  float* out = (float*)d_out;

  _Float16* ws = (_Float16*)d_ws;
  const size_t PROJ = (size_t)NROWS * 64;     // 8,388,608 halves per tensor
  _Float16* qp = ws;
  _Float16* kp = ws + PROJ;
  _Float16* vp = ws + 2 * PROJ;
  _Float16* ao = ws + 3 * PROJ;

  proj_qkv<<<dim3(NROWS / 128, 3), 256, 0, stream>>>(q, k, v, Wq, Wk, Wv, qp, kp, vp);
  flash_attn<<<dim3(L_ / 128, B_ * H_), 256, 0, stream>>>(qp, kp, vp, ao);
  out_proj<<<dim3(B_ * L_ / 16, E_ / 128), 256, 0, stream>>>(ao, Wo, bo, out);
}
